// sparse_linear_81398220193868
// MI455X (gfx1250) — matrix operand order repaired and checked by reading, not compiled, not run on hardware
//
#include <hip/hip_runtime.h>
#include <hip/hip_bf16.h>

typedef __attribute__((ext_vector_type(16))) _Float16 v16h;
typedef __attribute__((ext_vector_type(8)))  float    v8f;
typedef __attribute__((ext_vector_type(2)))  float    f32x2;
typedef __attribute__((ext_vector_type(4)))  unsigned v4u;
typedef __attribute__((ext_vector_type(8)))  int      v8i;
typedef __attribute__((ext_vector_type(4)))  int      v4i;

#define TM 128
#define TN 64
#define TK 32

#ifndef __has_builtin
#define __has_builtin(x) 0
#endif
#if __has_builtin(__builtin_amdgcn_tensor_load_to_lds) && \
    __has_builtin(__builtin_amdgcn_s_wait_tensorcnt)
#define HAVE_TDM 1
#else
#define HAVE_TDM 0
#endif

__device__ __forceinline__ unsigned pack2(_Float16 a, _Float16 b) {
    unsigned lo = (unsigned)__builtin_bit_cast(unsigned short, a);
    unsigned hi = (unsigned)__builtin_bit_cast(unsigned short, b);
    return lo | (hi << 16);
}

#if HAVE_TDM
// 1-D contiguous TDM copy: global -> LDS, n8 elements of 8 bytes.
// D# per CDNA5 ISA 8.3/8.4: group0 = {count|flags, lds_addr, gaddr_lo, gaddr_hi|type=2},
// group1 = data_size=8B, tensor_dim0=tile_dim0=n8, dims1.. = 1, stride0 = n8.
// This toolchain exposes the 6-arg builtin (g0 v4u, g1 v8i, g2 v4i, g3 v4i, g4 v8i, cpol).
__device__ __forceinline__ void tdm_load_1d(unsigned lds_addr, const void* gsrc, unsigned n8) {
    unsigned long long ga = (unsigned long long)(uintptr_t)gsrc;
    v4u g0;
    g0.x = 1u;                                                // count=1 (valid), no gather
    g0.y = lds_addr;                                          // LDS byte address
    g0.z = (unsigned)(ga & 0xffffffffu);                      // global addr [31:0]
    g0.w = (unsigned)((ga >> 32) & 0x01ffffffu) | (2u << 30); // addr[56:32] | type=2
    v8i g1;
    g1[0] = (int)(3u << 16);                                  // data_size = 8 bytes
    g1[1] = (int)((n8 & 0xffffu) << 16);                      // tensor_dim0[15:0]
    g1[2] = (int)((n8 >> 16) & 0xffffu) | (1 << 16);          // tensor_dim0[31:16] | tensor_dim1=1
    g1[3] = (int)((n8 & 0xffffu) << 16);                      // tile_dim0 = n8
    g1[4] = 1;                                                // tile_dim1 = 1
    g1[5] = (int)n8;                                          // tensor_dim0_stride[31:0]
    g1[6] = 0;
    g1[7] = 0;
    v4i z4 = {0, 0, 0, 0};
    v8i z8 = {0, 0, 0, 0, 0, 0, 0, 0};
    __builtin_amdgcn_tensor_load_to_lds(g0, g1, z4, z4, z8, 0);
}
#else
__device__ __forceinline__ void coop_copy16(int tid, uint4* dst, const uint4* src, int n16) {
    for (int i = tid; i < n16; i += 256) dst[i] = src[i];
}
#endif

// ---------------------------------------------------------------------------
// Path 1 prep A: split f32 activations into hi/lo f16 planes, stored directly
// in WMMA A-fragment order: Asw[K/32][M/16][2(hi,lo)][32 lanes][16 halves].
// A 16x32 f16 layout: lane<16 : M=lane,   K in {0..7, 16..23}
//                     lane>=16: M=lane-16, K in {8..15, 24..31}
// ---------------------------------------------------------------------------
__global__ void sl_asplit_kernel(const float* __restrict__ A,
                                 unsigned short* __restrict__ Asw,
                                 int M, int K) {
    const long long total = (long long)M * (K >> 1);
    const long long idx = (long long)blockIdx.x * blockDim.x + threadIdx.x;
    if (idx >= total) return;
    const int kp = (int)(idx % (K >> 1));
    const int m  = (int)(idx / (K >> 1));
    const int k  = kp * 2;                       // even
    const f32x2 a2 = *(const f32x2*)(A + (size_t)m * K + k);
    const _Float16 h0 = (_Float16)a2.x, h1 = (_Float16)a2.y;
    const _Float16 l0 = (_Float16)(a2.x - (float)h0);
    const _Float16 l1 = (_Float16)(a2.y - (float)h1);
    const int k0 = k >> 5, kk = k & 31, ms = m >> 4, ml = m & 15;
    const int g = kk >> 3;
    const int lane = ml + 16 * (g & 1);
    const int e    = kk - 8 * ((g + 1) >> 1);    // even
    const size_t base = ((((size_t)k0 * (M >> 4) + ms) * 2) * 32 + lane) * 16 + e; // halves
    ((unsigned*)Asw)[base >> 1]         = pack2(h0, h1);   // hi plane (p=0)
    ((unsigned*)Asw)[(base + 512) >> 1] = pack2(l0, l1);   // lo plane (p=1, +32*16 halves)
}

// ---------------------------------------------------------------------------
// Path 1 prep B: scatter CSR directly into WMMA B-fragment order:
// Wsw[K/32][N/16][2(hi,lo)][32 lanes][16 halves].
// B 32x16 f16 layout: lane<16 : N=lane, K in {0..7, 16..23} ; lane>=16: N=lane-16, K in {8..15, 24..31}
// ---------------------------------------------------------------------------
__global__ void sl_densify_sw_kernel(const int* __restrict__ row_index,
                                     const int* __restrict__ col_index,
                                     const float* __restrict__ values,
                                     unsigned short* __restrict__ Wsw,
                                     int H, int nnz) {
    const int h     = blockIdx.x;
    const int start = row_index[h];
    const int end   = (h + 1 < H) ? row_index[h + 1] : nnz;
    const int ns = h >> 4, nl = h & 15;
    for (int j = start + (int)threadIdx.x; j < end; j += (int)blockDim.x) {
        const int   c = col_index[j];
        const float v = values[j];
        const _Float16 vh = (_Float16)v;
        const _Float16 vl = (_Float16)(v - (float)vh);
        const int k0 = c >> 5, kk = c & 31;
        const int lane = nl + 16 * ((kk >> 3) & 1);
        const int e    = (kk & 7) + 8 * (kk >> 4);
        const size_t base = ((((size_t)k0 * (H >> 4) + ns) * 2) * 32 + lane) * 16 + e;
        Wsw[base]       = __builtin_bit_cast(unsigned short, vh);
        Wsw[base + 512] = __builtin_bit_cast(unsigned short, vl);
    }
}

// ---------------------------------------------------------------------------
// Path 1 GEMM: operands already fragment-ordered in global memory; main loop
// is TDM async copies (double-buffered LDS) + ds_load_b128 + v_wmma only.
// Split-f16: hi*hi + lo*hi + hi*lo, f32 accumulate. 12 WMMA / wave / K-step.
// ---------------------------------------------------------------------------
__global__ __launch_bounds__(256)
void sl_gemm_tdm_kernel(const unsigned short* __restrict__ Asw,
                        const unsigned short* __restrict__ Wsw,
                        const float* __restrict__ bias,
                        float* __restrict__ C,
                        int M, int N, int K) {
    __shared__ alignas(32) _Float16 aLDS[2][8 * 2 * 32 * 16]; // 2 x 16KB
    __shared__ alignas(32) _Float16 bLDS[2][4 * 2 * 32 * 16]; // 2 x  8KB

    const int tid  = (int)threadIdx.x;
    const int lane = tid & 31;
    const int wid  = tid >> 5;
    const int wm   = wid >> 1;   // 0..3
    const int wn   = wid & 1;    // 0..1

    const int m0 = (int)blockIdx.y * TM;
    const int n0 = (int)blockIdx.x * TN;
    const int steps = K / TK;
    const int Msub = M >> 4, Nsub = N >> 4;
    const int msub0 = m0 >> 4, nsub0 = n0 >> 4;

    v8f acc[2][2] = {};

#if HAVE_TDM
#define SL_ISSUE(kk, bb)                                                                     \
    do {                                                                                     \
        if (wid == 0) {                                                                      \
            tdm_load_1d((unsigned)(uintptr_t)&aLDS[(bb)][0],                                 \
                        (const char*)Asw + ((size_t)(kk) * Msub + msub0) * 2048, 2048);      \
            tdm_load_1d((unsigned)(uintptr_t)&bLDS[(bb)][0],                                 \
                        (const char*)Wsw + ((size_t)(kk) * Nsub + nsub0) * 2048, 1024);      \
        }                                                                                    \
    } while (0)
#else
#define SL_ISSUE(kk, bb)                                                                     \
    do {                                                                                     \
        coop_copy16(tid, (uint4*)&aLDS[(bb)][0],                                             \
                    (const uint4*)((const char*)Asw + ((size_t)(kk) * Msub + msub0) * 2048), \
                    1024);                                                                   \
        coop_copy16(tid, (uint4*)&bLDS[(bb)][0],                                             \
                    (const uint4*)((const char*)Wsw + ((size_t)(kk) * Nsub + nsub0) * 2048), \
                    512);                                                                    \
    } while (0)
#endif

    // Prologue: fill both buffers.
    SL_ISSUE(0, 0);
    if (steps > 1) SL_ISSUE(1, 1);

    for (int k0 = 0; k0 < steps; ++k0) {
        const int buf = k0 & 1;
#if HAVE_TDM
        if (wid == 0) {
            if (k0 < steps - 1) __builtin_amdgcn_s_wait_tensorcnt(2); // newest pair may fly
            else                __builtin_amdgcn_s_wait_tensorcnt(0);
        }
#endif
        __syncthreads();

        const v16h* aF = (const v16h*)&aLDS[buf][0];
        const v16h* bF = (const v16h*)&bLDS[buf][0];
        v16h af_h[2], af_l[2], bf_h[2], bf_l[2];
        #pragma unroll
        for (int mi = 0; mi < 2; ++mi) {
            const int ms = wm * 2 + mi;
            af_h[mi] = aF[(ms * 2 + 0) * 32 + lane];
            af_l[mi] = aF[(ms * 2 + 1) * 32 + lane];
        }
        #pragma unroll
        for (int ni = 0; ni < 2; ++ni) {
            const int ns = wn * 2 + ni;
            bf_h[ni] = bF[(ns * 2 + 0) * 32 + lane];
            bf_l[ni] = bF[(ns * 2 + 1) * 32 + lane];
        }
        #pragma unroll
        for (int mi = 0; mi < 2; ++mi)
            #pragma unroll
            for (int ni = 0; ni < 2; ++ni) {
                acc[mi][ni] = __builtin_amdgcn_wmma_f32_16x16x32_f16(
                    false, af_h[mi], false, bf_h[ni], (short)0, acc[mi][ni], false, false);
                acc[mi][ni] = __builtin_amdgcn_wmma_f32_16x16x32_f16(
                    false, af_l[mi], false, bf_h[ni], (short)0, acc[mi][ni], false, false);
                acc[mi][ni] = __builtin_amdgcn_wmma_f32_16x16x32_f16(
                    false, af_h[mi], false, bf_l[ni], (short)0, acc[mi][ni], false, false);
            }

        __syncthreads();  // all reads of buf done before it is refilled
        if (k0 + 2 < steps) SL_ISSUE(k0 + 2, buf);
    }
#undef SL_ISSUE

    // Epilogue: C layout VGPR r -> M = r + 8*(lane>=16), N = lane&15.
    const int row_half = lane >> 4;
    const int cl       = lane & 15;
    #pragma unroll
    for (int mi = 0; mi < 2; ++mi) {
        const int gm = m0 + (wm * 2 + mi) * 16 + row_half * 8;
        #pragma unroll
        for (int ni = 0; ni < 2; ++ni) {
            const int gn = n0 + (wn * 2 + ni) * 16 + cl;
            const float bb = bias[gn];
            #pragma unroll
            for (int r = 0; r < 8; ++r)
                C[(size_t)(gm + r) * N + gn] = acc[mi][ni][r] + bb;
        }
    }
}

// ---------------------------------------------------------------------------
// Path 2 (ws >= 64MB only): round-1 kernel — densify packed, convert in-GEMM.
// ---------------------------------------------------------------------------
__global__ void sl_densify_kernel(const int* __restrict__ row_index,
                                  const int* __restrict__ col_index,
                                  const float* __restrict__ values,
                                  unsigned* __restrict__ Wt,
                                  int H, int nnz) {
    const int h     = blockIdx.x;
    const int start = row_index[h];
    const int end   = (h + 1 < H) ? row_index[h + 1] : nnz;
    for (int j = start + (int)threadIdx.x; j < end; j += (int)blockDim.x) {
        const int   c = col_index[j];
        const float v = values[j];
        const _Float16 vh = (_Float16)v;
        const _Float16 vl = (_Float16)(v - (float)vh);
        Wt[(size_t)c * H + h] = pack2(vh, vl);
    }
}

__global__ __launch_bounds__(256)
void sl_spmm_wmma_kernel(const float* __restrict__ A,
                         const unsigned* __restrict__ Wt,
                         const float* __restrict__ bias,
                         float* __restrict__ C,
                         int M, int N, int K) {
    __shared__ v16h a_hi[TM / 16][32];
    __shared__ v16h a_lo[TM / 16][32];
    __shared__ v16h b_hi[TN / 16][32];
    __shared__ v16h b_lo[TN / 16][32];

    const int tid  = (int)threadIdx.x;
    const int lane = tid & 31;
    const int wid  = tid >> 5;
    const int wm   = wid >> 1;
    const int wn   = wid & 1;

    const int m0 = (int)blockIdx.y * TM;
    const int n0 = (int)blockIdx.x * TN;

    v8f acc[2][2] = {};

    const int a_mrow = tid >> 4;
    const int a_kk   = (tid & 15) * 2;
    const int a_g    = a_kk >> 3;
    const int a_lane = a_mrow + 16 * (a_g & 1);
    const int a_dw   = (a_kk - 8 * ((a_g + 1) >> 1)) >> 1;

    const int b_n    = tid & 63;
    const int b_g4   = tid >> 6;
    const int b_nsub = b_n >> 4;
    const int b_nl   = b_n & 15;

    for (int k0 = 0; k0 < K; k0 += TK) {
        __syncthreads();
        #pragma unroll
        for (int p = 0; p < TM / 16; ++p) {
            const int m = p * 16 + a_mrow;
            const f32x2 a2 = *(const f32x2*)(A + (size_t)(m0 + m) * K + (k0 + a_kk));
            const _Float16 h0 = (_Float16)a2.x;
            const _Float16 h1 = (_Float16)a2.y;
            const _Float16 l0 = (_Float16)(a2.x - (float)h0);
            const _Float16 l1 = (_Float16)(a2.y - (float)h1);
            ((unsigned*)&a_hi[p][a_lane])[a_dw] = pack2(h0, h1);
            ((unsigned*)&a_lo[p][a_lane])[a_dw] = pack2(l0, l1);
        }
        #pragma unroll
        for (int i = 0; i < 4; ++i) {
            const int kk  = (b_g4 * 4 + i) * 2;
            const unsigned w0 = Wt[(size_t)(k0 + kk)     * N + (n0 + b_n)];
            const unsigned w1 = Wt[(size_t)(k0 + kk + 1) * N + (n0 + b_n)];
            const int bl  = b_nl + 16 * ((kk >> 3) & 1);
            const int dw  = ((kk & 7) + 8 * (kk >> 4)) >> 1;
            ((unsigned*)&b_hi[b_nsub][bl])[dw] = (w0 & 0xFFFFu) | (w1 << 16);
            ((unsigned*)&b_lo[b_nsub][bl])[dw] = (w0 >> 16)     | (w1 & 0xFFFF0000u);
        }
        __syncthreads();

        v16h af_h[2], af_l[2], bf_h[2], bf_l[2];
        #pragma unroll
        for (int mi = 0; mi < 2; ++mi) {
            af_h[mi] = a_hi[wm * 2 + mi][lane];
            af_l[mi] = a_lo[wm * 2 + mi][lane];
        }
        #pragma unroll
        for (int ni = 0; ni < 2; ++ni) {
            bf_h[ni] = b_hi[wn * 2 + ni][lane];
            bf_l[ni] = b_lo[wn * 2 + ni][lane];
        }
        #pragma unroll
        for (int mi = 0; mi < 2; ++mi)
            #pragma unroll
            for (int ni = 0; ni < 2; ++ni) {
                acc[mi][ni] = __builtin_amdgcn_wmma_f32_16x16x32_f16(
                    false, af_h[mi], false, bf_h[ni], (short)0, acc[mi][ni], false, false);
                acc[mi][ni] = __builtin_amdgcn_wmma_f32_16x16x32_f16(
                    false, af_l[mi], false, bf_h[ni], (short)0, acc[mi][ni], false, false);
                acc[mi][ni] = __builtin_amdgcn_wmma_f32_16x16x32_f16(
                    false, af_h[mi], false, bf_l[ni], (short)0, acc[mi][ni], false, false);
            }
    }

    const int row_half = lane >> 4;
    const int cl       = lane & 15;
    #pragma unroll
    for (int mi = 0; mi < 2; ++mi) {
        const int gm = m0 + (wm * 2 + mi) * 16 + row_half * 8;
        #pragma unroll
        for (int ni = 0; ni < 2; ++ni) {
            const int gn = n0 + (wn * 2 + ni) * 16 + cl;
            const float bb = bias[gn];
            #pragma unroll
            for (int r = 0; r < 8; ++r)
                C[(size_t)(gm + r) * N + gn] = acc[mi][ni][r] + bb;
        }
    }
}

// ---------------------------------------------------------------------------
// Path 3: exact-f32 CSR gather fallback.
// ---------------------------------------------------------------------------
__global__ void sl_fallback_kernel(const float* __restrict__ data,
                                   const int* __restrict__ row_index,
                                   const int* __restrict__ col_index,
                                   const float* __restrict__ values,
                                   const float* __restrict__ bias,
                                   float* __restrict__ out,
                                   int H, int K, int nnz, long long total) {
    const long long idx = (long long)blockIdx.x * blockDim.x + threadIdx.x;
    if (idx >= total) return;
    const int h = (int)(idx % H);
    const int t = (int)(idx / H);
    const int start = row_index[h];
    const int end   = (h + 1 < H) ? row_index[h + 1] : nnz;
    const float* drow = data + (size_t)t * K;
    float acc = bias[h];
    for (int j = start; j < end; ++j) acc += values[j] * drow[col_index[j]];
    out[idx] = acc;
}

extern "C" void kernel_launch(void* const* d_in, const int* in_sizes, int n_in,
                              void* d_out, int out_size, void* d_ws, size_t ws_size,
                              hipStream_t stream) {
    const float* data      = (const float*)d_in[0];
    const int*   row_index = (const int*)d_in[2];
    const int*   col_index = (const int*)d_in[3];
    const float* values    = (const float*)d_in[4];
    const float* bias      = (const float*)d_in[5];
    float*       out       = (float*)d_out;

    const int H    = in_sizes[2];            // out_features (N)
    const int nnz  = in_sizes[4];
    const int ntok = out_size / H;           // tokens (M)
    const int K    = in_sizes[0] / ntok;     // in_features

    const size_t wsw_bytes = (size_t)K * (size_t)H * 4u;      // fragment-ordered W (hi+lo)
    const size_t asw_bytes = (size_t)ntok * (size_t)K * 4u;   // fragment-ordered A (hi+lo)
    const bool tiles_ok = (ntok % TM == 0) && (H % TN == 0) && (K % TK == 0) && (H % 16 == 0);

    if (tiles_ok && ws_size >= wsw_bytes + asw_bytes) {
        // Path 1: pre-swizzled operands + TDM/async-pipelined WMMA GEMM.
        unsigned short* Wsw = (unsigned short*)d_ws;
        unsigned short* Asw = (unsigned short*)((char*)d_ws + wsw_bytes);
        (void)hipMemsetAsync(Wsw, 0, wsw_bytes, stream);
        hipLaunchKernelGGL(sl_densify_sw_kernel, dim3(H), dim3(256), 0, stream,
                           row_index, col_index, values, Wsw, H, nnz);
        const long long half_elems = (long long)ntok * (K >> 1);
        const int ablocks = (int)((half_elems + 255) / 256);
        hipLaunchKernelGGL(sl_asplit_kernel, dim3(ablocks), dim3(256), 0, stream,
                           data, Asw, ntok, K);
        dim3 grid(H / TN, ntok / TM);
        hipLaunchKernelGGL(sl_gemm_tdm_kernel, grid, dim3(256), 0, stream,
                           Asw, Wsw, bias, out, ntok, H, K);
        return;
    }

    if (tiles_ok && ws_size >= wsw_bytes) {
        // Path 2: packed-W densify + in-GEMM split.
        unsigned* Wt = (unsigned*)d_ws;
        (void)hipMemsetAsync(Wt, 0, wsw_bytes, stream);
        hipLaunchKernelGGL(sl_densify_kernel, dim3(H), dim3(256), 0, stream,
                           row_index, col_index, values, Wt, H, nnz);
        dim3 grid(H / TN, ntok / TM);
        hipLaunchKernelGGL(sl_spmm_wmma_kernel, grid, dim3(256), 0, stream,
                           data, Wt, bias, out, ntok, H, K);
        return;
    }

    const long long total = (long long)ntok * H;
    const int blocks = (int)((total + 255) / 256);
    hipLaunchKernelGGL(sl_fallback_kernel, dim3(blocks), dim3(256), 0, stream,
                       data, row_index, col_index, values, bias, out,
                       H, K, nnz, total);
}
